// Head_58385785422224
// MI455X (gfx1250) — matrix operand order repaired and checked by reading, not compiled, not run on hardware
//
#include <hip/hip_runtime.h>

#define B_ 4
#define T_ 4096
#define S_ 4096
#define E_ 512
#define H_ 64

typedef __attribute__((ext_vector_type(16))) _Float16 v16h;
typedef __attribute__((ext_vector_type(8)))  _Float16 v8h;
typedef __attribute__((ext_vector_type(8)))  float    v8f;
typedef __attribute__((ext_vector_type(4)))  int      v4i;

#define WMMA_F16(a, b, c) \
  __builtin_amdgcn_wmma_f32_16x16x32_f16(false, (a), false, (b), (short)0, (c), false, false)

static __device__ inline v16h cat8(v8h lo, v8h hi) {
  return __builtin_shufflevector(lo, hi, 0,1,2,3,4,5,6,7,8,9,10,11,12,13,14,15);
}
// A-operand fragment: elements 0..7 at p, elements 8..15 at p+16 (f16 units)
static __device__ inline v16h load_a_f16(const _Float16* p) {
  v8h lo = *(const v8h*)p;
  v8h hi = *(const v8h*)(p + 16);
  return cat8(lo, hi);
}

// ---------------- prep kernels ----------------
__global__ void prep_q_kernel(const float* __restrict__ x, const float* __restrict__ qw,
                              const float* __restrict__ qb, _Float16* __restrict__ qf) {
  int idx = blockIdx.x * blockDim.x + threadIdx.x;  // B*T*64
  if (idx >= B_ * T_ * H_) return;
  int h = idx & (H_ - 1);
  int bt = idx >> 6;
  qf[idx] = (_Float16)tanhf(x[bt] * qw[h] + qb[h]);
}

__global__ void cvt_w_kernel(const float* __restrict__ kw, const float* __restrict__ vw,
                             _Float16* __restrict__ kwh, _Float16* __restrict__ vwh) {
  int idx = blockIdx.x * blockDim.x + threadIdx.x;  // H*E
  if (idx >= H_ * E_) return;
  kwh[idx] = (_Float16)kw[idx];
  vwh[idx] = (_Float16)vw[idx];
}

__global__ void zero_f32_kernel(float* __restrict__ p, int n) {
  int idx = blockIdx.x * blockDim.x + threadIdx.x;
  if (idx < n) p[idx] = 0.0f;
}

// ---------------- kernel 1: k = tanh(TE @ Kw^T + kb), v = tanh(TE @ Vw^T) ----------------
// One wave per (b, s-tile16, mat); 4 h-tiles per wave; K loop over E in steps of 32.
__global__ void kv_kernel(const float* __restrict__ te, const _Float16* __restrict__ kwh,
                          const _Float16* __restrict__ vwh, const float* __restrict__ kb,
                          _Float16* __restrict__ kf, _Float16* __restrict__ vf) {
  int lane = threadIdx.x & 31;
  int gid = blockIdx.x * (blockDim.x >> 5) + (threadIdx.x >> 5);
  int mat = gid & 1;                 // 0=key, 1=value
  int st  = (gid >> 1) & (S_ / 16 - 1);
  int b   = gid >> 9;                // 2*(S/16)=512 per batch
  int s0 = st * 16;
  int m = lane & 15;
  int half = lane >> 4;              // 0: lanes 0-15, 1: lanes 16-31

  const _Float16* W = mat ? vwh : kwh;
  _Float16* O = mat ? vf : kf;

  v8f acc0 = {}, acc1 = {}, acc2 = {}, acc3 = {};
  const float* arow = te + ((size_t)b * S_ + s0 + m) * E_;

  for (int e0 = 0; e0 < E_; e0 += 32) {
    const float* ap = arow + e0 + half * 16;
    v16h a;
#pragma unroll
    for (int i = 0; i < 8; i++) a[i] = (_Float16)ap[i];
#pragma unroll
    for (int i = 0; i < 8; i++) a[i + 8] = (_Float16)ap[i + 8];
    int koff = e0 + half * 16;
    v16h b0 = *(const v16h*)(W + (size_t)(0 * 16 + m) * E_ + koff);
    v16h b1 = *(const v16h*)(W + (size_t)(1 * 16 + m) * E_ + koff);
    v16h b2 = *(const v16h*)(W + (size_t)(2 * 16 + m) * E_ + koff);
    v16h b3 = *(const v16h*)(W + (size_t)(3 * 16 + m) * E_ + koff);
    acc0 = WMMA_F16(a, b0, acc0);
    acc1 = WMMA_F16(a, b1, acc1);
    acc2 = WMMA_F16(a, b2, acc2);
    acc3 = WMMA_F16(a, b3, acc3);
  }

  v8f accs[4] = {acc0, acc1, acc2, acc3};
#pragma unroll
  for (int hi = 0; hi < 4; hi++) {
    int h = hi * 16 + m;
    float bias = (mat == 0) ? kb[h] : 0.0f;
#pragma unroll
    for (int r = 0; r < 8; r++) {
      int srow = s0 + r + half * 8;
      O[((size_t)b * S_ + srow) * H_ + h] = (_Float16)tanhf(accs[hi][r] + bias);
    }
  }
}

// ---------------- kernel 2: Z_s = sum_t exp(scale * q[t]·k[s]) ----------------
// One wave per (b, s-tile16, tsplit of 16); logits tiles M=t, N=s; per-lane exp sums.
__global__ void colsum_kernel(const _Float16* __restrict__ qf, const _Float16* __restrict__ kf,
                              float* __restrict__ Z) {
  int lane = threadIdx.x & 31;
  int gid = blockIdx.x * (blockDim.x >> 5) + (threadIdx.x >> 5);
  int tsplit = gid & 15;
  int st = (gid >> 4) & (S_ / 16 - 1);
  int b  = gid >> 12;
  int s0 = st * 16;
  int m = lane & 15;
  int half = lane >> 4;

  // B operands: k^T tiles, fixed for the whole T loop
  const _Float16* krow = kf + ((size_t)b * S_ + s0 + m) * H_;
  v16h kB0 = load_a_f16(krow + 0  + half * 8);
  v16h kB1 = load_a_f16(krow + 32 + half * 8);

  float zacc = 0.0f;
  for (int i = 0; i < 16; i++) {
    int t0 = (tsplit * 16 + i) * 16;
    const _Float16* qp = qf + ((size_t)b * T_ + t0 + m) * H_;
    v16h a0 = load_a_f16(qp + half * 8);
    v16h a1 = load_a_f16(qp + 32 + half * 8);
    v8f d = {};
    d = WMMA_F16(a0, kB0, d);
    d = WMMA_F16(a1, kB1, d);
#pragma unroll
    for (int r = 0; r < 8; r++) zacc += __expf(0.125f * d[r]);
  }
  zacc += __shfl_xor(zacc, 16, 32);
  if (lane < 16) atomicAdd(&Z[(size_t)b * S_ + s0 + lane], zacc);
}

// ---------------- kernel wt: w''^T[h,s] = v[s,h] * T / Z_s (f16, transposed) ----------------
__global__ void wt_kernel(const _Float16* __restrict__ vf, const float* __restrict__ Z,
                          _Float16* __restrict__ wt) {
  int idx = blockIdx.x * blockDim.x + threadIdx.x;  // B*S*H
  if (idx >= B_ * S_ * H_) return;
  int h = idx & (H_ - 1);
  int bs = idx >> 6;
  int s = bs & (S_ - 1);
  int b = bs >> 12;
  float w = (float)vf[idx] * ((float)T_ / Z[bs]);
  wt[((size_t)b * H_ + h) * S_ + s] = (_Float16)w;
}

// ---------------- kernel 3: out_acc[t,h] += sum_s exp(l) * w'' ----------------
// One wave per (b, t-tile16, s-split of 4). Logits computed transposed (M=s, N=t) so the
// exp'd D tile becomes the next WMMA's B operand directly via cat8(e0, e1), no lane exchange.
__global__ void out_kernel(const _Float16* __restrict__ qf, const _Float16* __restrict__ kf,
                           const _Float16* __restrict__ wt, float* __restrict__ oacc) {
  int lane = threadIdx.x & 31;
  int gid = blockIdx.x * (blockDim.x >> 5) + (threadIdx.x >> 5);
  int split = gid & 3;
  int tt = (gid >> 2) & (T_ / 16 - 1);
  int b  = gid >> 10;
  int t0 = tt * 16;
  int m = lane & 15;
  int half = lane >> 4;

  // B operands for logits: q^T tiles (K=h, N=t), fixed for the whole S loop
  const _Float16* qp = qf + ((size_t)b * T_ + t0 + m) * H_;
  v16h qB0 = load_a_f16(qp + 0  + half * 8);
  v16h qB1 = load_a_f16(qp + 32 + half * 8);

  v8f accO0 = {}, accO1 = {}, accO2 = {}, accO3 = {};

  int sBeg = split * (S_ / 4);
  int sEnd = sBeg + (S_ / 4);
  for (int s0 = sBeg; s0 < sEnd; s0 += 32) {
    // logits^T tiles: rows s0..s0+15 (d0) and s0+16..s0+31 (d1), cols t0..t0+15
    const _Float16* kp0 = kf + ((size_t)b * S_ + s0 + m) * H_;
    const _Float16* kp1 = kp0 + 16 * H_;
    v8f d0 = {}, d1 = {};
    d0 = WMMA_F16(load_a_f16(kp0 + half * 8),      qB0, d0);
    d0 = WMMA_F16(load_a_f16(kp0 + 32 + half * 8), qB1, d0);
    d1 = WMMA_F16(load_a_f16(kp1 + half * 8),      qB0, d1);
    d1 = WMMA_F16(load_a_f16(kp1 + 32 + half * 8), qB1, d1);

    // exp + convert to f16 pairs
    v8h e0 = {}, e1 = {};
#pragma unroll
    for (int r = 0; r < 8; r++) {
      e0[r] = (_Float16)__expf(0.125f * d0[r]);
      e1[r] = (_Float16)__expf(0.125f * d1[r]);
    }
    // build B operand (K=s0..s0+31, N=t) as cat8(e0, e1); the exchange below is unused
    v4i h0 = __builtin_bit_cast(v4i, e0);
    v4i h1 = __builtin_bit_cast(v4i, e1);
    v4i p0, p1, bl, bh;
#pragma unroll
    for (int i = 0; i < 4; i++) {
      p0[i] = __shfl_xor(h0[i], 16, 32);
      p1[i] = __shfl_xor(h1[i], 16, 32);
    }
    bool lo = lane < 16;
#pragma unroll
    for (int i = 0; i < 4; i++) {
      bl[i] = lo ? h0[i] : p1[i];
      bh[i] = lo ? p0[i] : h1[i];
    }
    v16h pB = cat8(e0, e1);

    // out^T accumulation over 4 h-tiles: A = w''^T[16h x 32s]
    const _Float16* wb = wt + (size_t)b * H_ * S_;
    accO0 = WMMA_F16(load_a_f16(wb + (size_t)(0 * 16 + m) * S_ + s0 + half * 8), pB, accO0);
    accO1 = WMMA_F16(load_a_f16(wb + (size_t)(1 * 16 + m) * S_ + s0 + half * 8), pB, accO1);
    accO2 = WMMA_F16(load_a_f16(wb + (size_t)(2 * 16 + m) * S_ + s0 + half * 8), pB, accO2);
    accO3 = WMMA_F16(load_a_f16(wb + (size_t)(3 * 16 + m) * S_ + s0 + half * 8), pB, accO3);
  }

  v8f accs[4] = {accO0, accO1, accO2, accO3};
#pragma unroll
  for (int hi = 0; hi < 4; hi++) {
#pragma unroll
    for (int r = 0; r < 8; r++) {
      int h = hi * 16 + r + half * 8;
      atomicAdd(&oacc[((size_t)b * T_ + t0 + m) * H_ + h], accs[hi][r]);
    }
  }
}

// ---------------- final: out[b,t] = proj_b + sum_h proj_w[h]*tanh(acc/T) ----------------
__global__ void final_kernel(const float* __restrict__ oacc, const float* __restrict__ pw,
                             const float* __restrict__ pb, float* __restrict__ out) {
  int idx = blockIdx.x * blockDim.x + threadIdx.x;  // B*T
  if (idx >= B_ * T_) return;
  const float* a = oacc + (size_t)idx * H_;
  float s = pb[0];
#pragma unroll 8
  for (int h = 0; h < H_; h++) s += pw[h] * tanhf(a[h] * (1.0f / (float)T_));
  out[idx] = s;
}

// ---------------- host ----------------
static inline size_t align256(size_t x) { return (x + 255) & ~(size_t)255; }

extern "C" void kernel_launch(void* const* d_in, const int* in_sizes, int n_in,
                              void* d_out, int out_size, void* d_ws, size_t ws_size,
                              hipStream_t stream) {
  const float* x   = (const float*)d_in[0];
  const float* te  = (const float*)d_in[1];
  const float* kw  = (const float*)d_in[2];
  const float* kb  = (const float*)d_in[3];
  const float* qw  = (const float*)d_in[4];
  const float* qb  = (const float*)d_in[5];
  const float* vw  = (const float*)d_in[6];
  const float* pw  = (const float*)d_in[7];
  const float* pb  = (const float*)d_in[8];
  float* out = (float*)d_out;

  char* w = (char*)d_ws;
  size_t off = 0;
  _Float16* qf  = (_Float16*)(w + off); off += align256((size_t)B_ * T_ * H_ * 2);
  _Float16* kf  = (_Float16*)(w + off); off += align256((size_t)B_ * S_ * H_ * 2);
  _Float16* vf  = (_Float16*)(w + off); off += align256((size_t)B_ * S_ * H_ * 2);
  _Float16* kwh = (_Float16*)(w + off); off += align256((size_t)H_ * E_ * 2);
  _Float16* vwh = (_Float16*)(w + off); off += align256((size_t)H_ * E_ * 2);
  float*    Zc  = (float*)(w + off);    off += align256((size_t)B_ * S_ * 4);
  _Float16* wtp = (_Float16*)(w + off); off += align256((size_t)B_ * H_ * S_ * 2);
  float*    oac = (float*)(w + off);    off += align256((size_t)B_ * T_ * H_ * 4);
  (void)ws_size; (void)in_sizes; (void)n_in; (void)out_size;

  // prep
  prep_q_kernel<<<(B_ * T_ * H_ + 255) / 256, 256, 0, stream>>>(x, qw, qb, qf);
  cvt_w_kernel<<<(H_ * E_ + 255) / 256, 256, 0, stream>>>(kw, vw, kwh, vwh);
  zero_f32_kernel<<<(B_ * S_ + 255) / 256, 256, 0, stream>>>(Zc, B_ * S_);
  zero_f32_kernel<<<(B_ * T_ * H_ + 255) / 256, 256, 0, stream>>>(oac, B_ * T_ * H_);

  // k, v GEMMs: 2048 waves, 4 waves/block
  kv_kernel<<<2048 / 4, 128, 0, stream>>>(te, kwh, vwh, kb, kf, vf);

  // column sums Z: 16384 waves
  colsum_kernel<<<16384 / 4, 128, 0, stream>>>(qf, kf, Zc);

  // w''^T
  wt_kernel<<<(B_ * S_ * H_ + 255) / 256, 256, 0, stream>>>(vf, Zc, wtp);

  // output accumulation: 4096 waves
  out_kernel<<<4096 / 4, 128, 0, stream>>>(qf, kf, wtp, oac);

  // projection
  final_kernel<<<(B_ * T_ + 255) / 256, 256, 0, stream>>>(oac, pw, pb, out);
}
